// MDDformerImproved_7722351198482
// MI455X (gfx1250) — compile-verified, matrix operand order checked by reading, not run on hardware
//
#include <hip/hip_runtime.h>
#include <math.h>

// MDDformer fused block for gfx1250 (MI455X): bf16 WMMA everywhere.
// B=8 T=1024 D=128 H=8 DH=16 FF=512

typedef __attribute__((ext_vector_type(16))) __bf16 v16bf;
typedef __attribute__((ext_vector_type(8)))  float  v8f;
typedef unsigned short u16;
typedef unsigned int   u32;

union Frag { v16bf v; u32 u[8]; };

__device__ __forceinline__ u16 f2bf(float f) {
  u32 x = __float_as_uint(f);
  u32 r = x + 0x7fffu + ((x >> 16) & 1u);   // round-to-nearest-even
  return (u16)(r >> 16);
}

// K index for half h=2v or 2v+1 of a 16-bit A/B fragment (ISA 7.12.2 layout)
__device__ __forceinline__ int kb_of(int v, int hi) {
  return ((v >> 2) << 4) + (hi << 3) + ((v & 3) << 1);
}

// ---------------------------------------------------------------------------
// Generic bf16 WMMA GEMM core: block computes 64 x 128 tile of C = A[M,K]*W[K,N]
// 8 waves arranged 2x4, each wave owns a 32x32 sub-tile (2x2 WMMA tiles).
// A row-major [M,K] bf16, W row-major [K,N] bf16. LDS row stride padded to 34.
// ---------------------------------------------------------------------------
__device__ __forceinline__ void gemm_core(
    const u16* __restrict__ A, const u16* __restrict__ W,
    int K, int N, int m0, int n0,
    u16* As, u16* Wt, v8f acc[2][2])
{
  const int tid = threadIdx.x;
  const int lane = tid & 31;
  const int w = tid >> 5;
  const int wr = w >> 2, wc = w & 3;
  const int lg = lane & 15, hi = lane >> 4;
#pragma unroll
  for (int i = 0; i < 2; ++i)
#pragma unroll
    for (int j = 0; j < 2; ++j)
#pragma unroll
      for (int r = 0; r < 8; ++r) acc[i][j][r] = 0.0f;

  for (int k0 = 0; k0 < K; k0 += 32) {
    __syncthreads();
    // stage A tile 64x32 (dword copies, 4/thread)
#pragma unroll
    for (int s = 0; s < 4; ++s) {
      int u = tid + 256 * s;
      int i = u >> 4, j = (u & 15) * 2;
      *(u32*)(As + i * 34 + j) =
          *(const u32*)(A + (size_t)(m0 + i) * K + k0 + j);
    }
    // stage W tile 32x128 transposed -> Wt[n][k]
#pragma unroll
    for (int s = 0; s < 8; ++s) {
      int u = tid + 256 * s;
      int k = u >> 6, n = (u & 63) * 2;
      u32 val = *(const u32*)(W + (size_t)(k0 + k) * N + n0 + n);
      Wt[n * 34 + k]       = (u16)(val & 0xffffu);
      Wt[(n + 1) * 34 + k] = (u16)(val >> 16);
    }
    if (k0 + 32 < K)   // software prefetch of the next A tile (global_prefetch)
      __builtin_prefetch(A + (size_t)(m0 + (tid >> 2)) * K + k0 + 32, 0, 1);
    __syncthreads();

    Frag af[2], bf2[2];
#pragma unroll
    for (int t = 0; t < 2; ++t) {
      const int r0 = wr * 32 + t * 16;
      const int c0 = wc * 32 + t * 16;
#pragma unroll
      for (int v = 0; v < 8; ++v) {
        const int kb = kb_of(v, hi);
        af[t].u[v]  = *(const u32*)(As + (r0 + lg) * 34 + kb);
        bf2[t].u[v] = *(const u32*)(Wt + (c0 + lg) * 34 + kb);
      }
    }
#pragma unroll
    for (int ti = 0; ti < 2; ++ti)
#pragma unroll
      for (int tj = 0; tj < 2; ++tj)
        acc[ti][tj] = __builtin_amdgcn_wmma_f32_16x16x32_bf16(
            false, af[ti].v, false, bf2[tj].v, (short)0, acc[ti][tj],
            false, false);
  }
}

// ---------------------------------------------------------------------------
// LayerNorm: one wave (32 lanes) per 128-wide row; two streams per launch.
// Writes optional f32 copy (residual path) and bf16 copy (GEMM path).
// ---------------------------------------------------------------------------
__global__ void __launch_bounds__(256) ln_kernel(
    const float* __restrict__ srcX, const float* __restrict__ srcY,
    const float* __restrict__ gX, const float* __restrict__ bX,
    const float* __restrict__ gY, const float* __restrict__ bY,
    float* __restrict__ outFX, float* __restrict__ outFY,
    u16* __restrict__ outBX, u16* __restrict__ outBY)
{
  const int w = threadIdx.x >> 5, lane = threadIdx.x & 31;
  const int row = blockIdx.x * 8 + w;
  const float *src, *g, *bb;
  float* oF; u16* oB;
  if (row < 8 * 1024) {
    src = srcX + (size_t)row * 128; g = gX; bb = bX;
    oF = outFX ? outFX + (size_t)row * 128 : nullptr;
    oB = outBX + (size_t)row * 128;
  } else {
    const int r2 = row - 8 * 1024;
    src = srcY + (size_t)r2 * 128; g = gY; bb = bY;
    oF = outFY ? outFY + (size_t)r2 * 128 : nullptr;
    oB = outBY + (size_t)r2 * 128;
  }
  const int c = lane * 4;
  float4 vv = *(const float4*)(src + c);
  float s  = vv.x + vv.y + vv.z + vv.w;
  float sq = vv.x*vv.x + vv.y*vv.y + vv.z*vv.z + vv.w*vv.w;
#pragma unroll
  for (int off = 1; off < 32; off <<= 1) {
    s  += __shfl_xor(s, off);
    sq += __shfl_xor(sq, off);
  }
  const float mean = s * (1.0f / 128.0f);
  const float var  = sq * (1.0f / 128.0f) - mean * mean;
  const float rstd = rsqrtf(var + 1e-5f);
  const float o0 = (vv.x - mean) * rstd * g[c + 0] + bb[c + 0];
  const float o1 = (vv.y - mean) * rstd * g[c + 1] + bb[c + 1];
  const float o2 = (vv.z - mean) * rstd * g[c + 2] + bb[c + 2];
  const float o3 = (vv.w - mean) * rstd * g[c + 3] + bb[c + 3];
  if (oF) { float4 ov = {o0, o1, o2, o3}; *(float4*)(oF + c) = ov; }
  *(u32*)(oB + c)     = (u32)f2bf(o0) | ((u32)f2bf(o1) << 16);
  *(u32*)(oB + c + 2) = (u32)f2bf(o2) | ((u32)f2bf(o3) << 16);
}

// ---------------------------------------------------------------------------
// Fused QKV: grid.z selects 6 GEMM jobs (shared weights, two streams).
// Q/K scattered into concat layouts [B,H,T,32] (qy|qx, kx|ky); V -> [B,H,T,16].
// ---------------------------------------------------------------------------
__global__ void __launch_bounds__(256) qkv_kernel(
    const u16* __restrict__ xn, const u16* __restrict__ yn,
    const u16* __restrict__ wq, const u16* __restrict__ wk,
    const u16* __restrict__ wv,
    u16* __restrict__ qc, u16* __restrict__ kc,
    u16* __restrict__ vx, u16* __restrict__ vy)
{
  __shared__ __attribute__((aligned(16))) u16 As[64 * 34];
  __shared__ __attribute__((aligned(16))) u16 Wt[128 * 34];
  const u16 *A, *W; u16* dst; int off = 0; bool isV = false;
  switch (blockIdx.z) {
    case 0:  A = yn; W = wq; dst = qc; off = 0;  break;
    case 1:  A = xn; W = wq; dst = qc; off = 16; break;
    case 2:  A = xn; W = wk; dst = kc; off = 0;  break;
    case 3:  A = yn; W = wk; dst = kc; off = 16; break;
    case 4:  A = xn; W = wv; dst = vx; isV = true; break;
    default: A = yn; W = wv; dst = vy; isV = true; break;
  }
  v8f acc[2][2];
  const int m0 = blockIdx.x * 64;
  gemm_core(A, W, 128, 128, m0, 0, As, Wt, acc);

  const int lane = threadIdx.x & 31, w = threadIdx.x >> 5;
  const int wr = w >> 2, wc = w & 3, lg = lane & 15, hi = lane >> 4;
#pragma unroll
  for (int ti = 0; ti < 2; ++ti)
#pragma unroll
    for (int tj = 0; tj < 2; ++tj)
#pragma unroll
      for (int r = 0; r < 8; ++r) {
        const int tok = m0 + wr * 32 + ti * 16 + hi * 8 + r;
        const int n = wc * 32 + tj * 16 + lg;
        const int b = tok >> 10, t = tok & 1023;
        const int h = n >> 4, d = n & 15;
        const size_t bh = (size_t)(b * 8 + h) * 1024 + t;
        const size_t idx = isV ? bh * 16 + d : bh * 32 + off + d;
        dst[idx] = f2bf(acc[ti][tj][r]);
      }
}

// ---------------------------------------------------------------------------
// Flash attention. Score: single K=32 WMMA on [qy|qx]x[kx|ky]^T == averaged
// dual score map (x0.25 for the 0.5 mean and x... total scale 0.5/sqrt(16)).
// Online softmax per 16-row q tile per wave; P re-laid-out via wave LDS.
// ---------------------------------------------------------------------------
__global__ void __launch_bounds__(256) attn_kernel(
    const u16* __restrict__ qc, const u16* __restrict__ kc,
    const u16* __restrict__ vx, const u16* __restrict__ vy,
    u16* __restrict__ a1, u16* __restrict__ a2)
{
  __shared__ __attribute__((aligned(16))) u16 Klds[32 * 34];
  __shared__ __attribute__((aligned(16))) u16 Vxt[16 * 34];
  __shared__ __attribute__((aligned(16))) u16 Vyt[16 * 34];
  __shared__ __attribute__((aligned(16))) u16 Pl[8][16 * 34];
  const int tid = threadIdx.x, lane = tid & 31, w = tid >> 5;
  const int lg = lane & 15, hi = lane >> 4;
  const int bh = blockIdx.y;
  const int b = bh >> 3, h = bh & 7;
  const int q0 = blockIdx.x * 128 + w * 16;
  const size_t baseQK = (size_t)bh * 1024 * 32;
  const size_t baseV  = (size_t)bh * 1024 * 16;

  Frag qf;
#pragma unroll
  for (int v = 0; v < 8; ++v)
    qf.u[v] = *(const u32*)(qc + baseQK + (size_t)(q0 + lg) * 32 + kb_of(v, hi));

  float mrow[8], lrow[8];
  v8f acc1, acc2;
#pragma unroll
  for (int r = 0; r < 8; ++r) { mrow[r] = -3.0e38f; lrow[r] = 0.f; acc1[r] = 0.f; acc2[r] = 0.f; }
  u16* Pme = Pl[w];

  for (int k0 = 0; k0 < 1024; k0 += 32) {
    __syncthreads();
#pragma unroll
    for (int s = 0; s < 2; ++s) {              // K-cat tile: 32 keys x 32 dims
      const int u = tid + 256 * s;
      const int key = u >> 4, j = (u & 15) * 2;
      *(u32*)(Klds + key * 34 + j) =
          *(const u32*)(kc + baseQK + (size_t)(k0 + key) * 32 + j);
    }
    {                                          // V tiles transposed: Vt[d][key]
      const int key = tid >> 3, d = (tid & 7) * 2;
      u32 vvx = *(const u32*)(vx + baseV + (size_t)(k0 + key) * 16 + d);
      Vxt[d * 34 + key]       = (u16)(vvx & 0xffffu);
      Vxt[(d + 1) * 34 + key] = (u16)(vvx >> 16);
      u32 vvy = *(const u32*)(vy + baseV + (size_t)(k0 + key) * 16 + d);
      Vyt[d * 34 + key]       = (u16)(vvy & 0xffffu);
      Vyt[(d + 1) * 34 + key] = (u16)(vvy >> 16);
    }
    __syncthreads();

    Frag bk0, bk1;
#pragma unroll
    for (int v = 0; v < 8; ++v) {
      const int kb = kb_of(v, hi);
      bk0.u[v] = *(const u32*)(Klds + lg * 34 + kb);
      bk1.u[v] = *(const u32*)(Klds + (16 + lg) * 34 + kb);
    }
    v8f s0, s1;
#pragma unroll
    for (int r = 0; r < 8; ++r) { s0[r] = 0.f; s1[r] = 0.f; }
    s0 = __builtin_amdgcn_wmma_f32_16x16x32_bf16(false, qf.v, false, bk0.v, (short)0, s0, false, false);
    s1 = __builtin_amdgcn_wmma_f32_16x16x32_bf16(false, qf.v, false, bk1.v, (short)0, s1, false, false);

#pragma unroll
    for (int r = 0; r < 8; ++r) {              // online softmax per row
      const float e0 = s0[r] * 0.125f, e1 = s1[r] * 0.125f;   // 0.5/sqrt(16)
      float rm = fmaxf(e0, e1);
      rm = fmaxf(rm, __shfl_xor(rm, 1));
      rm = fmaxf(rm, __shfl_xor(rm, 2));
      rm = fmaxf(rm, __shfl_xor(rm, 4));
      rm = fmaxf(rm, __shfl_xor(rm, 8));
      const float mnew  = fmaxf(mrow[r], rm);
      const float alpha = __expf(mrow[r] - mnew);
      const float p0 = __expf(e0 - mnew), p1 = __expf(e1 - mnew);
      float rs = p0 + p1;
      rs += __shfl_xor(rs, 1);
      rs += __shfl_xor(rs, 2);
      rs += __shfl_xor(rs, 4);
      rs += __shfl_xor(rs, 8);
      lrow[r] = lrow[r] * alpha + rs;
      mrow[r] = mnew;
      acc1[r] *= alpha;
      acc2[r] *= alpha;
      const int m = hi * 8 + r;                // C-layout row -> P scratch
      Pme[m * 34 + lg]      = f2bf(p0);
      Pme[m * 34 + 16 + lg] = f2bf(p1);
    }
    asm volatile("s_wait_dscnt 0" ::: "memory");  // LDS store->load, same wave

    Frag pf, fvx, fvy;
#pragma unroll
    for (int v = 0; v < 8; ++v) {
      const int kb = kb_of(v, hi);
      pf.u[v]  = *(const u32*)(Pme + lg * 34 + kb);
      fvx.u[v] = *(const u32*)(Vxt + lg * 34 + kb);
      fvy.u[v] = *(const u32*)(Vyt + lg * 34 + kb);
    }
    acc1 = __builtin_amdgcn_wmma_f32_16x16x32_bf16(false, pf.v, false, fvx.v, (short)0, acc1, false, false);
    acc2 = __builtin_amdgcn_wmma_f32_16x16x32_bf16(false, pf.v, false, fvy.v, (short)0, acc2, false, false);
  }
#pragma unroll
  for (int r = 0; r < 8; ++r) {
    const int tok = b * 1024 + q0 + hi * 8 + r;
    const int col = h * 16 + lg;
    const float inv = 1.0f / lrow[r];
    a1[(size_t)tok * 128 + col] = f2bf(acc1[r] * inv);
    a2[(size_t)tok * 128 + col] = f2bf(acc2[r] * inv);
  }
}

// ---------------------------------------------------------------------------
// Output projection + bias + residual (xn/yn), f32 out. grid.z = stream.
// ---------------------------------------------------------------------------
__global__ void __launch_bounds__(256) oproj_kernel(
    const u16* __restrict__ a1, const u16* __restrict__ a2,
    const u16* __restrict__ wox, const u16* __restrict__ woy,
    const float* __restrict__ box, const float* __restrict__ boy,
    const float* __restrict__ xn, const float* __restrict__ yn,
    float* __restrict__ x1, float* __restrict__ y1)
{
  __shared__ __attribute__((aligned(16))) u16 As[64 * 34];
  __shared__ __attribute__((aligned(16))) u16 Wt[128 * 34];
  const u16 *A, *W; const float *bias, *res; float* out;
  if (blockIdx.z == 0) { A = a1; W = wox; bias = box; res = xn; out = x1; }
  else                 { A = a2; W = woy; bias = boy; res = yn; out = y1; }
  v8f acc[2][2];
  const int m0 = blockIdx.x * 64;
  gemm_core(A, W, 128, 128, m0, 0, As, Wt, acc);
  const int lane = threadIdx.x & 31, w = threadIdx.x >> 5;
  const int wr = w >> 2, wc = w & 3, lg = lane & 15, hi = lane >> 4;
#pragma unroll
  for (int ti = 0; ti < 2; ++ti)
#pragma unroll
    for (int tj = 0; tj < 2; ++tj)
#pragma unroll
      for (int r = 0; r < 8; ++r) {
        const int tok = m0 + wr * 32 + ti * 16 + hi * 8 + r;
        const int n = wc * 32 + tj * 16 + lg;
        out[(size_t)tok * 128 + n] =
            acc[ti][tj][r] + bias[n] + res[(size_t)tok * 128 + n];
      }
}

// FFN up-projection + exact GELU -> bf16 hidden. grid: (M/64, N/128, stream)
__global__ void __launch_bounds__(256) ffn1_kernel(
    const u16* __restrict__ x1n, const u16* __restrict__ y1n,
    const u16* __restrict__ w1x, const u16* __restrict__ w1y,
    const float* __restrict__ b1x, const float* __restrict__ b1y,
    u16* __restrict__ hx, u16* __restrict__ hy)
{
  __shared__ __attribute__((aligned(16))) u16 As[64 * 34];
  __shared__ __attribute__((aligned(16))) u16 Wt[128 * 34];
  const u16 *A, *W; const float* bias; u16* out;
  if (blockIdx.z == 0) { A = x1n; W = w1x; bias = b1x; out = hx; }
  else                 { A = y1n; W = w1y; bias = b1y; out = hy; }
  v8f acc[2][2];
  const int m0 = blockIdx.x * 64;
  const int n0 = blockIdx.y * 128;
  gemm_core(A, W, 128, 512, m0, n0, As, Wt, acc);
  const int lane = threadIdx.x & 31, w = threadIdx.x >> 5;
  const int wr = w >> 2, wc = w & 3, lg = lane & 15, hi = lane >> 4;
#pragma unroll
  for (int ti = 0; ti < 2; ++ti)
#pragma unroll
    for (int tj = 0; tj < 2; ++tj)
#pragma unroll
      for (int r = 0; r < 8; ++r) {
        const int tok = m0 + wr * 32 + ti * 16 + hi * 8 + r;
        const int n = n0 + wc * 32 + tj * 16 + lg;
        const float z = acc[ti][tj][r] + bias[n];
        const float g = 0.5f * z * (1.0f + erff(z * 0.70710678118654752f));
        out[(size_t)tok * 512 + n] = f2bf(g);
      }
}

// FFN down-projection + bias + residual -> final f32 output (x2 then y2).
__global__ void __launch_bounds__(256) ffn2_kernel(
    const u16* __restrict__ hx, const u16* __restrict__ hy,
    const u16* __restrict__ w2x, const u16* __restrict__ w2y,
    const float* __restrict__ b2x, const float* __restrict__ b2y,
    const float* __restrict__ x1, const float* __restrict__ y1,
    float* __restrict__ dout)
{
  __shared__ __attribute__((aligned(16))) u16 As[64 * 34];
  __shared__ __attribute__((aligned(16))) u16 Wt[128 * 34];
  const u16 *A, *W; const float *bias, *res; float* out;
  if (blockIdx.z == 0) { A = hx; W = w2x; bias = b2x; res = x1; out = dout; }
  else { A = hy; W = w2y; bias = b2y; res = y1; out = dout + (size_t)8192 * 128; }
  v8f acc[2][2];
  const int m0 = blockIdx.x * 64;
  gemm_core(A, W, 512, 128, m0, 0, As, Wt, acc);
  const int lane = threadIdx.x & 31, w = threadIdx.x >> 5;
  const int wr = w >> 2, wc = w & 3, lg = lane & 15, hi = lane >> 4;
#pragma unroll
  for (int ti = 0; ti < 2; ++ti)
#pragma unroll
    for (int tj = 0; tj < 2; ++tj)
#pragma unroll
      for (int r = 0; r < 8; ++r) {
        const int tok = m0 + wr * 32 + ti * 16 + hi * 8 + r;
        const int n = wc * 32 + tj * 16 + lg;
        out[(size_t)tok * 128 + n] =
            acc[ti][tj][r] + bias[n] + res[(size_t)tok * 128 + n];
      }
}

__global__ void cvt_kernel(const float* __restrict__ src, u16* __restrict__ dst, int n) {
  const int i = blockIdx.x * 256 + threadIdx.x;
  if (i < n) dst[i] = f2bf(src[i]);
}

// ---------------------------------------------------------------------------
extern "C" void kernel_launch(void* const* d_in, const int* in_sizes, int n_in,
                              void* d_out, int out_size, void* d_ws, size_t ws_size,
                              hipStream_t stream) {
  (void)in_sizes; (void)n_in; (void)out_size; (void)ws_size;
  const float* x     = (const float*)d_in[0];
  const float* y     = (const float*)d_in[1];
  const float* Wq    = (const float*)d_in[2];
  const float* Wk    = (const float*)d_in[3];
  const float* Wv    = (const float*)d_in[4];
  const float* Wox   = (const float*)d_in[5];
  const float* box   = (const float*)d_in[6];
  const float* Woy   = (const float*)d_in[7];
  const float* boy   = (const float*)d_in[8];
  const float* ln1xg = (const float*)d_in[9];
  const float* ln1xb = (const float*)d_in[10];
  const float* ln1yg = (const float*)d_in[11];
  const float* ln1yb = (const float*)d_in[12];
  const float* ln2xg = (const float*)d_in[13];
  const float* ln2xb = (const float*)d_in[14];
  const float* ln2yg = (const float*)d_in[15];
  const float* ln2yb = (const float*)d_in[16];
  const float* Wx1   = (const float*)d_in[17];
  const float* bx1   = (const float*)d_in[18];
  const float* Wx2   = (const float*)d_in[19];
  const float* bx2   = (const float*)d_in[20];
  const float* Wy1   = (const float*)d_in[21];
  const float* by1   = (const float*)d_in[22];
  const float* Wy2   = (const float*)d_in[23];
  const float* by2   = (const float*)d_in[24];

  unsigned char* ws = (unsigned char*)d_ws;
  const size_t ND = (size_t)8192 * 128;    // B*T*D
  float* xnF = (float*)(ws);
  float* ynF = (float*)(ws + ND * 4);
  float* x1F = (float*)(ws + 2 * ND * 4);
  float* y1F = (float*)(ws + 3 * ND * 4);
  u16* p = (u16*)(ws + 4 * ND * 4);
  u16* xnB = p; p += ND;
  u16* ynB = p; p += ND;
  u16* QC  = p; p += 2 * ND;   // [B,H,T,32] = qy|qx
  u16* KC  = p; p += 2 * ND;   // [B,H,T,32] = kx|ky
  u16* VX  = p; p += ND;       // [B,H,T,16]
  u16* VY  = p; p += ND;
  u16* A1  = p; p += ND;       // [B*T,128]
  u16* A2  = p; p += ND;
  u16* X1N = p; p += ND;
  u16* Y1N = p; p += ND;
  u16* HX  = p; p += 4 * ND;   // [B*T,512]
  u16* HY  = p; p += 4 * ND;
  u16* WqB  = p; p += 128 * 128;
  u16* WkB  = p; p += 128 * 128;
  u16* WvB  = p; p += 128 * 128;
  u16* WoxB = p; p += 128 * 128;
  u16* WoyB = p; p += 128 * 128;
  u16* Wx1B = p; p += 128 * 512;
  u16* Wy1B = p; p += 128 * 512;
  u16* Wx2B = p; p += 512 * 128;
  u16* Wy2B = p; p += 512 * 128;

  // weight conversions f32 -> bf16 (cheap; deterministic every call)
  cvt_kernel<<<64,  256, 0, stream>>>(Wq,  WqB,  128 * 128);
  cvt_kernel<<<64,  256, 0, stream>>>(Wk,  WkB,  128 * 128);
  cvt_kernel<<<64,  256, 0, stream>>>(Wv,  WvB,  128 * 128);
  cvt_kernel<<<64,  256, 0, stream>>>(Wox, WoxB, 128 * 128);
  cvt_kernel<<<64,  256, 0, stream>>>(Woy, WoyB, 128 * 128);
  cvt_kernel<<<256, 256, 0, stream>>>(Wx1, Wx1B, 128 * 512);
  cvt_kernel<<<256, 256, 0, stream>>>(Wy1, Wy1B, 128 * 512);
  cvt_kernel<<<256, 256, 0, stream>>>(Wx2, Wx2B, 512 * 128);
  cvt_kernel<<<256, 256, 0, stream>>>(Wy2, Wy2B, 512 * 128);

  // 1) LN1 on x,y -> f32 + bf16
  ln_kernel<<<2048, 256, 0, stream>>>(x, y, ln1xg, ln1xb, ln1yg, ln1yb,
                                      xnF, ynF, xnB, ynB);
  // 2) fused QKV (6 GEMMs)
  qkv_kernel<<<dim3(128, 1, 6), 256, 0, stream>>>(xnB, ynB, WqB, WkB, WvB,
                                                  QC, KC, VX, VY);
  // 3) dual-value flash attention
  attn_kernel<<<dim3(8, 64), 256, 0, stream>>>(QC, KC, VX, VY, A1, A2);
  // 4) output projection + residual
  oproj_kernel<<<dim3(128, 1, 2), 256, 0, stream>>>(A1, A2, WoxB, WoyB,
                                                    box, boy, xnF, ynF, x1F, y1F);
  // 5) LN2 -> bf16 only
  ln_kernel<<<2048, 256, 0, stream>>>(x1F, y1F, ln2xg, ln2xb, ln2yg, ln2yb,
                                      nullptr, nullptr, X1N, Y1N);
  // 6) FFN up + GELU
  ffn1_kernel<<<dim3(128, 4, 2), 256, 0, stream>>>(X1N, Y1N, Wx1B, Wy1B,
                                                   bx1, by1, HX, HY);
  // 7) FFN down + residual -> d_out (x2 then y2)
  ffn2_kernel<<<dim3(128, 1, 2), 256, 0, stream>>>(HX, HY, Wx2B, Wy2B,
                                                   bx2, by2, x1F, y1F,
                                                   (float*)d_out);
}
